// Encoder_57990648430823
// MI455X (gfx1250) — matrix operand order repaired and checked by reading, not compiled, not run on hardware
//
#include <hip/hip_runtime.h>
#include <hip/hip_bf16.h>
#include <math.h>

// ---------------------------------------------------------------------------
// MI455X (gfx1250) implementation.
//
// Input leaf order assumption (setup_inputs dict insertion order, params
// flattened like jax.tree_util: dict keys sorted, lists in order):
//   d_in[0]            x                      (8192*24*16*2 f32)
//   enc4 block (base=1):  bn_b[0..2](16), bn_g[0..2](16), fc_W(4x128),
//                         fc_b(4), sl_W[l][s] 12 mats, sl_b[l][s] 12 vecs
//   enc5 block (base=33): bn_b[0..2](8),  bn_g[0..2](8),  fc_W(5x160),
//                         fc_b(5), sl_W 12, sl_b 12
//   d_in[65] codebook_2, d_in[66] codebook_3   (unused: levels are analytic)
// ---------------------------------------------------------------------------

typedef __bf16 bf16_t;
typedef __attribute__((ext_vector_type(16))) __bf16 v16bf;
typedef __attribute__((ext_vector_type(8)))  __bf16 v8bf;
typedef __attribute__((ext_vector_type(8)))  float  v8f;
typedef __attribute__((ext_vector_type(4))) unsigned int u32x4;
typedef __attribute__((ext_vector_type(8))) int          i32x8;
typedef __attribute__((ext_vector_type(4))) int          i32x4;

#define DEV __device__ __forceinline__

DEV float mishf(float x) {
  float sp = (x > 20.f) ? x : log1pf(expf(x));
  return x * tanhf(sp);
}

DEV bool is2bit(int i) {   // BIT2_IDX = {102,103,88,29,18,25}
  return (i == 18) || (i == 25) || (i == 29) || (i == 88) || (i == 102) || (i == 103);
}

// ---------------------------------------------------------------------------
// Kernel 1: per-batch variance, stable descending argsort of 24 channels,
// gather (x-0.5) into bf16 GEMM-A buffers X5 (B*8 x 32) and X4 (B*16 x 32).
// ---------------------------------------------------------------------------
__global__ __launch_bounds__(256)
void prep_sort_gather(const float* __restrict__ x,
                      bf16_t* __restrict__ X5, bf16_t* __restrict__ X4) {
  __shared__ float xv[8][24];
  __shared__ int   ord[8][24];
  const int lane = threadIdx.x & 31;
  const int w    = threadIdx.x >> 5;
  const int b    = blockIdx.x * 8 + w;
  const float* xb = x + (size_t)b * 768;

  if (lane < 24) {
    float s = 0.f;
    for (int f = 0; f < 32; ++f) { float d = xb[lane * 32 + f] - 0.5f; s += d * d; }
    xv[w][lane] = s;   // 1/32 scale irrelevant for argsort
  }
  __syncthreads();
  if (lane == 0) {
    int tmp[24];
    for (int i = 0; i < 24; ++i) tmp[i] = i;
    for (int i = 1; i < 24; ++i) {         // stable insertion sort, descending
      int ti = tmp[i]; float key = xv[w][ti]; int j = i - 1;
      while (j >= 0 && xv[w][tmp[j]] < key) { tmp[j + 1] = tmp[j]; --j; }
      tmp[j + 1] = ti;
    }
    for (int i = 0; i < 24; ++i) ord[w][i] = tmp[i];
  }
  __syncthreads();
  for (int k = 0; k < 24; ++k) {
    int ch = ord[w][k];
    float v = xb[ch * 32 + lane] - 0.5f;
    if (k < 8) X5[((size_t)b * 8 + k) * 32 + lane] = (bf16_t)v;
    else       X4[((size_t)b * 16 + (k - 8)) * 32 + lane] = (bf16_t)v;
  }
}

// ---------------------------------------------------------------------------
// Kernel 2: weight matrix (tout,tin) f32 -> bf16 padded (Np32, Kp), zero pad.
// ---------------------------------------------------------------------------
__global__ void conv_weight(const float* __restrict__ W, int tout, int tin,
                            bf16_t* __restrict__ Wp, int Kpad, size_t total) {
  size_t i = (size_t)blockIdx.x * blockDim.x + threadIdx.x;
  if (i >= total) return;
  int n = (int)(i / Kpad), k = (int)(i % Kpad);
  float v = (n < tout && k < tin) ? W[(size_t)n * tin + k] : 0.f;
  Wp[i] = (bf16_t)v;
}

// ---------------------------------------------------------------------------
// Kernel 3: WMMA GEMM + fused bias + mish + split scatter.
// Block = 256 thr (8 waves) -> 128 rows x 32 cols. W panel (32 x Kpad bf16)
// staged in LDS once per block via the Tensor Data Mover in a k-chunked
// layout [kstep][row][k%32] (conflict-free ds_load_b128 reads).
// Each wave owns one 16-row strip and both 16-col tiles (2 accumulators).
// ---------------------------------------------------------------------------
#define MAX_KPAD 416

__global__ __launch_bounds__(256)
void gemm_mish_split(const bf16_t* __restrict__ X, int ldx,
                     const bf16_t* __restrict__ Wp, int Kpad,
                     const float* __restrict__ bias, int Nvalid, int o,
                     bf16_t* __restrict__ cat, int ldcat, int outOff,
                     bf16_t* __restrict__ chain, int ldchain) {
  __shared__ __align__(16) bf16_t sB[32 * MAX_KPAD];
  const int lane = threadIdx.x & 31;
  const int wave = threadIdx.x >> 5;
  const int half = lane >> 4;
  const int l16  = lane & 15;
  const int m0 = blockIdx.x * 128 + wave * 16;
  const int n0 = blockIdx.y * 32;

  // ---- stage B panel: W rows [n0, n0+32) in [kstep][row][k%32] order ------
#if __has_builtin(__builtin_amdgcn_tensor_load_to_lds)
  if (wave == 0) {
    // Tensor DMA descriptor (see CDNA5 ISA):
    // view W panel as 3D tensor x=32 elems, y=32 rows (stride Kpad),
    // z=Kpad/32 chunks (stride 32); TDM fills LDS contiguously in x,y,z.
    unsigned long long ga =
        (unsigned long long)(uintptr_t)(const void*)(Wp + (size_t)n0 * Kpad);
    unsigned int ldsOff = (unsigned int)(uintptr_t)(void*)sB;
    u32x4 g0;
    g0[0] = 1u;                                            // count=1
    g0[1] = ldsOff;                                        // lds_addr (bytes)
    g0[2] = (unsigned int)ga;                              // global_addr lo
    g0[3] = (unsigned int)((ga >> 32) & 0x01FFFFFFull) | (2u << 30); // hi|type=2
    const int zc = Kpad >> 5;                              // kstep count
    i32x8 g1;
    g1[0] = 1 << 16;                 // data_size=1 (2 bytes), mask=0, no flags
    g1[1] = 32 << 16;                // tensor_dim0 = 32  (bits 79:48 lo)
    g1[2] = 32 << 16;                // tensor_dim0 hi=0 | tensor_dim1 = 32
    g1[3] = 32 << 16;                // tensor_dim1 hi=0 | tile_dim0 = 32
    g1[4] = 32 | (zc << 16);         // tile_dim1 = 32 | tile_dim2 = zc
    g1[5] = Kpad;                    // tensor_dim0_stride lo (row stride)
    g1[6] = 32 << 16;                // td0_stride hi=0 | tensor_dim1_stride = 32
    g1[7] = 0;                       // tensor_dim1_stride hi = 0
    i32x4 g2; g2[0] = zc; g2[1] = 0; g2[2] = 0; g2[3] = 0;  // tensor_dim2 = zc
    i32x4 g3; g3[0] = 0;  g3[1] = 0; g3[2] = 0; g3[3] = 0;
    i32x8 g4; g4[0] = 0;  g4[1] = 0; g4[2] = 0; g4[3] = 0;
              g4[4] = 0;  g4[5] = 0; g4[6] = 0; g4[7] = 0;
    // 6-arg form (clang-23 / therock-10.0 lane)
    __builtin_amdgcn_tensor_load_to_lds(g0, g1, g2, g3, g4, 0);
    __builtin_amdgcn_s_wait_tensorcnt(0);
  }
#else
  {
    const bf16_t* src = Wp + (size_t)n0 * Kpad;
    const int total = 32 * Kpad;
    for (int i = threadIdx.x * 8; i < total; i += 256 * 8) {
      int xo = i & 31, n = (i >> 5) & 31, z = i >> 10;
      *(v8bf*)(sB + i) = *(const v8bf*)(src + (size_t)n * Kpad + z * 32 + xo);
    }
  }
#endif
  __syncthreads();

  v8f acc0 = {}, acc1 = {};
  const bf16_t* xrow = X + (size_t)(m0 + l16) * ldx;
  const int boff = l16 * 32 + 8 * half;    // within 1K-elem kstep block

  for (int k0 = 0; k0 < Kpad; k0 += 32) {
    // A fragment (16-bit A 16x32): lane half h holds K in [8h,8h+8)U[16+8h,..)
    v8bf a0 = *(const v8bf*)(xrow + k0 + 8 * half);
    v8bf a1 = *(const v8bf*)(xrow + k0 + 16 + 8 * half);
    v16bf a = __builtin_shufflevector(a0, a1, 0,1,2,3,4,5,6,7,8,9,10,11,12,13,14,15);
    // B fragments from LDS panel: rows 0..15 (tile 0), rows 16..31 (tile 1)
    const bf16_t* kb = sB + (k0 << 5);     // (k0/32)*1024 elements
    v8bf p0 = *(const v8bf*)(kb + boff);
    v8bf p1 = *(const v8bf*)(kb + boff + 16);
    v16bf bA = __builtin_shufflevector(p0, p1, 0,1,2,3,4,5,6,7,8,9,10,11,12,13,14,15);
    v8bf q0 = *(const v8bf*)(kb + 512 + boff);
    v8bf q1 = *(const v8bf*)(kb + 512 + boff + 16);
    v16bf bB = __builtin_shufflevector(q0, q1, 0,1,2,3,4,5,6,7,8,9,10,11,12,13,14,15);
    acc0 = __builtin_amdgcn_wmma_f32_16x16x32_bf16(false, a, false, bA,
                                                   (short)0, acc0, false, false);
    acc1 = __builtin_amdgcn_wmma_f32_16x16x32_bf16(false, a, false, bB,
                                                   (short)0, acc1, false, false);
  }

  auto emit = [&](const v8f& acc, int col) {
    if (col < Nvalid) {
      const float bv = bias[col];
#pragma unroll
      for (int r = 0; r < 8; ++r) {
        int row = m0 + r + 8 * half;   // C/D: lane -> N, vgpr r -> M = r+8*half
        float t = mishf(acc[r] + bv);
        if (col < o) cat[(size_t)row * ldcat + outOff + col] = (bf16_t)mishf(t);
        else         chain[(size_t)row * ldchain + (col - o)] = (bf16_t)t;
      }
    }
  };
  emit(acc0, n0 + l16);
  emit(acc1, n0 + 16 + l16);
}

// ---------------------------------------------------------------------------
// Kernel 4: zero pad columns [from, ld) of a row-major bf16 buffer.
// ---------------------------------------------------------------------------
__global__ void zero_cols(bf16_t* __restrict__ buf, int ld, int from, size_t total) {
  size_t i = (size_t)blockIdx.x * blockDim.x + threadIdx.x;
  if (i >= total) return;
  int padw = ld - from;
  size_t r = i / padw;
  int c = from + (int)(i % padw);
  buf[r * (size_t)ld + c] = (bf16_t)0.f;
}

__global__ void zero_f32(float* p, int n) {
  int i = blockIdx.x * blockDim.x + threadIdx.x;
  if (i < n) p[i] = 0.f;
}

// ---------------------------------------------------------------------------
// Kernel 5: batchnorm statistics (channel c = row % num), LDS tree reduce.
// ---------------------------------------------------------------------------
__global__ __launch_bounds__(256)
void bn_stats(const bf16_t* __restrict__ C, int ldc, int Wv, int num,
              int rowsPerChan, int chunk, float* __restrict__ sums) {
  const int c = blockIdx.x;
  int r0 = blockIdx.y * chunk;
  int r1 = r0 + chunk; if (r1 > rowsPerChan) r1 = rowsPerChan;
  float s = 0.f, s2 = 0.f;
  for (int rr = r0; rr < r1; ++rr) {
    const bf16_t* row = C + ((size_t)rr * num + c) * ldc;
    for (int col = threadIdx.x; col < Wv; col += 256) {
      float v = (float)row[col];
      s += v; s2 += v * v;
    }
  }
  __shared__ float sh[256], sh2[256];
  sh[threadIdx.x] = s; sh2[threadIdx.x] = s2;
  __syncthreads();
  for (int st = 128; st > 0; st >>= 1) {
    if (threadIdx.x < st) { sh[threadIdx.x] += sh[threadIdx.x + st]; sh2[threadIdx.x] += sh2[threadIdx.x + st]; }
    __syncthreads();
  }
  if (threadIdx.x == 0) {
    atomicAdd(&sums[2 * c],     sh[0]);
    atomicAdd(&sums[2 * c + 1], sh2[0]);
  }
}

// ---------------------------------------------------------------------------
// Kernel 6: batchnorm apply -> bf16 A-matrix for next layer (zero K-pad cols).
// ---------------------------------------------------------------------------
__global__ void bn_apply(const bf16_t* __restrict__ C, int ldc, int Wv, int num,
                         const float* __restrict__ sums, const float* __restrict__ g,
                         const float* __restrict__ bt, float cntInv,
                         bf16_t* __restrict__ Xo, int ldo, size_t total) {
  size_t i = (size_t)blockIdx.x * blockDim.x + threadIdx.x;
  if (i >= total) return;
  size_t r = i / ldo;
  int col = (int)(i % ldo);
  float y = 0.f;
  if (col < Wv) {
    int c = (int)(r % num);
    float m   = sums[2 * c] * cntInv;
    float var = sums[2 * c + 1] * cntInv - m * m;
    float v = (float)C[r * (size_t)ldc + col];
    y = (v - m) * rsqrtf(var + 1e-5f) * g[c] + bt[c];
  }
  Xo[r * (size_t)ldo + col] = (bf16_t)y;
}

// ---------------------------------------------------------------------------
// Kernel 7: final FC (N = fb <= 5) + sigmoid + analytic VQ quantize.
// d_out = [strite(B,104) | quant_out(B,104) | out(B,104)], strite == quant.
// ---------------------------------------------------------------------------
__global__ __launch_bounds__(256)
void fc_sig_quant(const bf16_t* __restrict__ X, int K,
                  const float* __restrict__ W, const float* __restrict__ bias,
                  int fb, int num, int outBase, float* __restrict__ out, int M) {
  int r = blockIdx.x * blockDim.x + threadIdx.x;
  if (r >= M) return;
  int b = r / num, c = r % num;
  float acc[5];
  for (int j = 0; j < fb; ++j) acc[j] = bias[j];
  const bf16_t* xr = X + (size_t)r * K;
  for (int k = 0; k < K; ++k) {
    float xv = (float)xr[k];
    for (int j = 0; j < fb; ++j) acc[j] += xv * W[j * K + k];
  }
  const size_t R = (size_t)8192 * 104;
  for (int j = 0; j < fb; ++j) {
    float v = 1.f / (1.f + expf(-acc[j]));
    int i = outBase + c * fb + j;
    int L = is2bit(i) ? 4 : 8;
    int kq = (int)floorf(v * (float)L);
    kq = kq < 0 ? 0 : (kq > L - 1 ? L - 1 : kq);
    float q = ((float)kq + 0.5f) / (float)L;
    size_t oi = (size_t)b * 104 + i;
    out[oi] = q;            // strite
    out[R + oi] = q;        // quant_out
    out[2 * R + oi] = v;    // out (pre-quant sigmoid)
  }
}

// ---------------------------------------------------------------------------
// Host orchestration
// ---------------------------------------------------------------------------
struct SubCfg { int K, N, o; };

// [enc][layer][sub]: enc 0 = fb5 (num=8), enc 1 = fb4 (num=16)
static const SubCfg SUBS[2][3][4] = {
  { // enc5: chain (32,320),(320,400),(400,160)
    { {32, 320, 128}, {192, 192,  96}, { 96,  96, 64}, {32, 32, 32} },
    { {320, 400, 160}, {240, 240, 120}, {120, 120, 80}, {40, 40, 40} },
    { {400, 304,  64}, {240, 168,  48}, {120,  72, 32}, {40, 16, 16} },
  },
  { // enc4: chain (32,256),(256,320),(320,128)
    { {32, 255, 102}, {153, 152,  76}, { 76,  76, 51}, {25, 27, 27} },
    { {256, 320, 128}, {192, 192,  96}, { 96,  96, 64}, {32, 32, 32} },
    { {320, 243,  51}, {192, 134,  38}, { 96,  57, 25}, {32, 14, 14} },
  }
};
static const int OC[2][3]  = { {320, 400, 160}, {255, 320, 128} };
static const int MROWS[2]  = { 65536, 131072 };
static const int NUMC[2]   = { 8, 16 };
static const int FBV[2]    = { 5, 4 };

static inline int up32(int v) { return (v + 31) & ~31; }

extern "C" void kernel_launch(void* const* d_in, const int* in_sizes, int n_in,
                              void* d_out, int out_size, void* d_ws, size_t ws_size,
                              hipStream_t stream) {
  (void)in_sizes; (void)n_in; (void)out_size; (void)ws_size;
  const float* x = (const float*)d_in[0];
  char* ws = (char*)d_ws;

  // workspace layout (256B aligned)
  size_t off = 0;
  auto alloc = [&](size_t bytes) { size_t o = off; off = (off + bytes + 255) & ~(size_t)255; return o; };
  const size_t catBytes = (size_t)131072 * 416;   // covers 65536*416 & 131072*320
  size_t oX5   = alloc((size_t)65536  * 32 * 2);
  size_t oX4   = alloc((size_t)131072 * 32 * 2);
  size_t oChA  = alloc((size_t)131072 * 256 * 2);
  size_t oChB  = alloc((size_t)131072 * 256 * 2);
  size_t oCat  = alloc(catBytes * 2);
  size_t oXbn  = alloc(catBytes * 2);
  size_t oSums = alloc(64 * sizeof(float));
  size_t woff[2][3][4];
  size_t wtot = 0;
  for (int e = 0; e < 2; ++e)
    for (int l = 0; l < 3; ++l)
      for (int s = 0; s < 4; ++s) {
        woff[e][l][s] = wtot;
        wtot += (size_t)up32(SUBS[e][l][s].N) * up32(SUBS[e][l][s].K);
      }
  size_t oW = alloc(wtot * 2);

  bf16_t* X5  = (bf16_t*)(ws + oX5);
  bf16_t* X4  = (bf16_t*)(ws + oX4);
  bf16_t* chA = (bf16_t*)(ws + oChA);
  bf16_t* chB = (bf16_t*)(ws + oChB);
  bf16_t* cat = (bf16_t*)(ws + oCat);
  bf16_t* xbn = (bf16_t*)(ws + oXbn);
  float*  sums = (float*)(ws + oSums);
  bf16_t* wbase = (bf16_t*)(ws + oW);

  // leaf index helpers (enc 0 = fb5 -> base 33; enc 1 = fb4 -> base 1)
  auto base  = [&](int e) { return (e == 0) ? 33 : 1; };
  auto slW   = [&](int e, int l, int s) { return (const float*)d_in[base(e) + 8  + l * 4 + s]; };
  auto slB   = [&](int e, int l, int s) { return (const float*)d_in[base(e) + 20 + l * 4 + s]; };
  auto bnB   = [&](int e, int l) { return (const float*)d_in[base(e) + l]; };
  auto bnG   = [&](int e, int l) { return (const float*)d_in[base(e) + 3 + l]; };
  auto fcW   = [&](int e) { return (const float*)d_in[base(e) + 6]; };
  auto fcB   = [&](int e) { return (const float*)d_in[base(e) + 7]; };

  // 1) weight conversion (f32 -> padded bf16, zero padding)
  for (int e = 0; e < 2; ++e)
    for (int l = 0; l < 3; ++l)
      for (int s = 0; s < 4; ++s) {
        const SubCfg& cfg = SUBS[e][l][s];
        int Kp = up32(cfg.K), Np = up32(cfg.N);
        size_t total = (size_t)Np * Kp;
        conv_weight<<<(unsigned)((total + 255) / 256), 256, 0, stream>>>(
            slW(e, l, s), cfg.N, cfg.K, wbase + woff[e][l][s], Kp, total);
      }

  // 2) variance sort + gather into bf16 A buffers
  prep_sort_gather<<<1024, 256, 0, stream>>>(x, X5, X4);

  // 3) encoders
  for (int e = 0; e < 2; ++e) {
    const int M = MROWS[e], num = NUMC[e];
    const bf16_t* Xin = (e == 0) ? X5 : X4;
    int ldin = 32;
    for (int l = 0; l < 3; ++l) {
      const int oc = OC[e][l];
      const int ldcat = up32(oc);
      int outOff = 0;
      bf16_t* chainIn = nullptr;
      for (int s = 0; s < 4; ++s) {
        const SubCfg& cfg = SUBS[e][l][s];
        int Kp = up32(cfg.K), Np = up32(cfg.N);
        const bf16_t* Xs; int ldx;
        if (s == 0) { Xs = Xin; ldx = ldin; }
        else {
          Xs = chainIn; ldx = Kp;
          if (Kp > cfg.K) {   // zero stale pad columns before GEMM reads them
            size_t zt = (size_t)M * (Kp - cfg.K);
            zero_cols<<<(unsigned)((zt + 255) / 256), 256, 0, stream>>>(chainIn, Kp, cfg.K, zt);
          }
        }
        bf16_t* chainOut; int ldch;
        if (s < 3) {
          chainOut = (s == 0) ? chA : ((s == 1) ? chB : chA);  // ping-pong
          ldch = up32(SUBS[e][l][s + 1].K);
        } else {
          chainOut = cat; ldch = ldcat;   // never written: o == N
        }
        dim3 grid((unsigned)(M / 128), (unsigned)(Np / 32));
        gemm_mish_split<<<grid, 256, 0, stream>>>(
            Xs, ldx, wbase + woff[e][l][s], Kp, slB(e, l, s), cfg.N, cfg.o,
            cat, ldcat, outOff, chainOut, ldch);
        outOff += cfg.o;
        chainIn = chainOut;
      }
      // batchnorm over (batch, features) per channel
      zero_f32<<<1, 64, 0, stream>>>(sums, 64);
      const int rowsPerChan = M / num;           // == 8192
      bn_stats<<<dim3((unsigned)num, (unsigned)(rowsPerChan / 256)), 256, 0, stream>>>(
          cat, ldcat, oc, num, rowsPerChan, 256, sums);
      const float cntInv = 1.f / ((float)rowsPerChan * (float)oc);
      const int ldo = up32(oc);
      size_t total = (size_t)M * ldo;
      bn_apply<<<(unsigned)((total + 255) / 256), 256, 0, stream>>>(
          cat, ldcat, oc, num, sums, bnG(e, l), bnB(e, l), cntInv, xbn, ldo, total);
      Xin = xbn; ldin = ldo;
    }
    // FC head + sigmoid + quantize, writes all three output regions
    fc_sig_quant<<<(unsigned)((M + 255) / 256), 256, 0, stream>>>(
        xbn, (e == 0) ? 160 : 128, fcW(e), fcB(e), FBV[e], num,
        (e == 0) ? 0 : 40, (float*)d_out, M);
  }
}
